// GraphTransformerLayer_56573309223521
// MI455X (gfx1250) — matrix operand order repaired and checked by reading, not compiled, not run on hardware
//
#include <hip/hip_runtime.h>
#include <hip/hip_bf16.h>
#include <math.h>

#define NN 50000
#define EE 800000
#define CC 128
#define HH 8
#define HD 16
#define C4 512   // 4*C

typedef _Float16 h16;
typedef __attribute__((ext_vector_type(16))) _Float16 v16h;
typedef __attribute__((ext_vector_type(8)))  _Float16 v8h;
typedef __attribute__((ext_vector_type(8)))  float    v8f;

#define WMMA_F16(a, b, c) \
  __builtin_amdgcn_wmma_f32_16x16x32_f16(false, (a), false, (b), (short)0, (c), false, false)

// ---------------- WMMA fragment loaders (ISA 7.12.2 layouts, wave32) -------
// A: 16x32 f16 from row-major f16 matrix. Two contiguous 16B runs per lane.
__device__ __forceinline__ v16h load_a_f16(const h16* __restrict__ A, int row, int ld,
                                           int kb, int hi) {
  const h16* p = A + row * ld + kb + 8 * hi;
  v8h lo = *(const v8h*)(p);
  v8h up = *(const v8h*)(p + 16);
  v16h a;
#pragma unroll
  for (int e = 0; e < 8; ++e) { a[e] = lo[e]; a[8 + e] = up[e]; }
  return a;
}
// A from f32 source (only needed for the atomic-accumulated hout).
__device__ __forceinline__ v16h load_a_f32(const float* __restrict__ A, int row, int ld,
                                           int kb, int hi) {
  v16h a;
  const float* p = A + row * ld + kb + 8 * hi;
#pragma unroll
  for (int e = 0; e < 8; ++e) { a[e] = (h16)p[e]; a[8 + e] = (h16)p[16 + e]; }
  return a;
}
// B: pre-packed fragment layout: [kchunk][tile][lane][16 halves] contiguous.
__device__ __forceinline__ v16h load_b_packed(const h16* __restrict__ Wp, int kc, int tile,
                                              int numTiles, int lane) {
  const h16* p = Wp + ((size_t)((kc * numTiles + tile) * 32 + lane)) * 16;
  v8h lo = *(const v8h*)(p);
  v8h up = *(const v8h*)(p + 8);
  v16h b;
#pragma unroll
  for (int e = 0; e < 8; ++e) { b[e] = lo[e]; b[8 + e] = up[e]; }
  return b;
}

// --------------- order-preserving float<->uint keys for atomicMax ----------
__device__ __forceinline__ unsigned fkey(float f) {
  unsigned u = __float_as_uint(f);
  return (u & 0x80000000u) ? ~u : (u | 0x80000000u);
}
__device__ __forceinline__ float finv(unsigned k) {
  unsigned u = (k & 0x80000000u) ? (k ^ 0x80000000u) : ~k;
  return __uint_as_float(u);
}

// ============== 0a) pack f32 weight [K,N] -> f16 WMMA fragment layout ======
__global__ void pack_weight_kernel(const float* __restrict__ W, h16* __restrict__ Wp,
                                   int K, int N) {
  int idx = blockIdx.x * 256 + threadIdx.x;       // one packed half per thread
  if (idx >= K * N) return;
  int e    = idx & 15;
  int l    = (idx >> 4) & 31;
  int rest = idx >> 9;
  int numTiles = N >> 4;
  int t  = rest % numTiles;
  int kc = rest / numTiles;
  int krow = kc * 32 + 8 * (l >> 4) + (e & 7) + 16 * (e >> 3);
  int col  = t * 16 + (l & 15);
  Wp[idx] = (h16)W[krow * N + col];
}

// ============== 0b) x f32 -> f16 (GEMM operand copy) =======================
__global__ void cvt_x_kernel(const float* __restrict__ x, h16* __restrict__ xh) {
  int idx = blockIdx.x * 256 + threadIdx.x;
  if (idx < NN * CC) xh[idx] = (h16)x[idx];
}

// ======================= 1) fused QKV projection ===========================
__global__ __launch_bounds__(256) void qkv_kernel(
    const h16* __restrict__ xh,
    const h16* __restrict__ Wq, const float* __restrict__ bq,
    const h16* __restrict__ Wk, const float* __restrict__ bk,
    const h16* __restrict__ Wv, const float* __restrict__ bv,
    float* __restrict__ q, float* __restrict__ k, float* __restrict__ v) {
  const int wave = threadIdx.x >> 5, lane = threadIdx.x & 31;
  const int hi = lane >> 4, l15 = lane & 15;
  const int rowB = blockIdx.x * 16;
  v8f cq = {}, ck = {}, cv = {};
#pragma unroll
  for (int kc = 0; kc < CC / 32; ++kc) {
    v16h a  = load_a_f16(xh, rowB + l15, CC, kc * 32, hi);
    v16h b0 = load_b_packed(Wq, kc, wave, CC / 16, lane);
    v16h b1 = load_b_packed(Wk, kc, wave, CC / 16, lane);
    v16h b2 = load_b_packed(Wv, kc, wave, CC / 16, lane);
    cq = WMMA_F16(a, b0, cq);
    ck = WMMA_F16(a, b1, ck);
    cv = WMMA_F16(a, b2, cv);
  }
  const int c = wave * 16 + l15;
#pragma unroll
  for (int i = 0; i < 8; ++i) {
    int r = rowB + i + 8 * hi;
    q[r * CC + c] = cq[i] + bq[c];
    k[r * CC + c] = ck[i] + bk[c];
    v[r * CC + c] = cv[i] + bv[c];
  }
}

// ======================= 2) init accumulators ==============================
__global__ void init_kernel(float* __restrict__ hout, unsigned* __restrict__ rmax,
                            float* __restrict__ rsum) {
  int idx = blockIdx.x * 256 + threadIdx.x;
  if (idx < NN * CC) hout[idx] = 0.f;
  if (idx < HH) { rmax[idx] = 0u; rsum[idx] = 0.f; }
}

// ======================= 3) edge scores + global max =======================
__global__ __launch_bounds__(256) void scores_kernel(
    const float* __restrict__ q, const float* __restrict__ k,
    const int* __restrict__ ei, float* __restrict__ scores,
    unsigned* __restrict__ rmax) {
  __shared__ unsigned smax[HH];
  if (threadIdx.x < HH) smax[threadIdx.x] = 0u;
  __syncthreads();
  const int e = blockIdx.x * 256 + threadIdx.x;
  const int row = ei[e];                      // faithful: both q and k by row
  const float4* qr = (const float4*)(q + row * CC);
  const float4* kr = (const float4*)(k + row * CC);
#pragma unroll
  for (int h = 0; h < HH; ++h) {
    float d = 0.f;
#pragma unroll
    for (int i = 0; i < 4; ++i) {
      float4 qq = qr[h * 4 + i], kk = kr[h * 4 + i];
      d += qq.x * kk.x + qq.y * kk.y + qq.z * kk.z + qq.w * kk.w;
    }
    d *= 0.25f;                               // 1/sqrt(16)
    scores[e * HH + h] = d;
    atomicMax(&smax[h], fkey(d));
  }
  __syncthreads();
  if (threadIdx.x < HH) atomicMax(&rmax[threadIdx.x], smax[threadIdx.x]);
}

// ======================= 4) global exp-sum per head ========================
__global__ __launch_bounds__(256) void expsum_kernel(
    const float* __restrict__ scores, const unsigned* __restrict__ rmax,
    float* __restrict__ rsum) {
  __shared__ float ssum[HH];
  if (threadIdx.x < HH) ssum[threadIdx.x] = 0.f;
  __syncthreads();
  const int e = blockIdx.x * 256 + threadIdx.x;
#pragma unroll
  for (int h = 0; h < HH; ++h) {
    float m = finv(rmax[h]);
    atomicAdd(&ssum[h], expf(scores[e * HH + h] - m));
  }
  __syncthreads();
  if (threadIdx.x < HH) atomicAdd(&rsum[threadIdx.x], ssum[threadIdx.x]);
}

// ======================= 5) weighted scatter (segment_sum) =================
__global__ __launch_bounds__(256) void scatter_kernel(
    const float* __restrict__ v, const float* __restrict__ scores,
    const int* __restrict__ ei, const unsigned* __restrict__ rmax,
    const float* __restrict__ rsum, float* __restrict__ hout) {
  const int e = blockIdx.x * 256 + threadIdx.x;
  const int row = ei[e];
  const int col = ei[EE + e];
  float attn[HH];
#pragma unroll
  for (int h = 0; h < HH; ++h)
    attn[h] = expf(scores[e * HH + h] - finv(rmax[h])) / rsum[h];
  const float4* vr = (const float4*)(v + col * CC);
  float* dst = hout + row * CC;
#pragma unroll 8
  for (int i = 0; i < CC / 4; ++i) {
    float4 vv = vr[i];
    float a = attn[(i * 4) >> 4];
    atomicAdd(&dst[i * 4 + 0], vv.x * a);
    atomicAdd(&dst[i * 4 + 1], vv.y * a);
    atomicAdd(&dst[i * 4 + 2], vv.z * a);
    atomicAdd(&dst[i * 4 + 3], vv.w * a);
  }
}

// ========== 6) O-projection + residual + LN1 + LN2 (WMMA + LDS) ============
__global__ __launch_bounds__(256) void oproj_ln_kernel(
    const float* __restrict__ hout, const h16* __restrict__ Wo,
    const float* __restrict__ bo, const float* __restrict__ x,
    const float* __restrict__ s1, const float* __restrict__ b1,
    const float* __restrict__ s2, const float* __restrict__ b2,
    float* __restrict__ h, h16* __restrict__ g) {
  __shared__ float tile[16][CC + 1];
  const int wave = threadIdx.x >> 5, lane = threadIdx.x & 31;
  const int hi = lane >> 4, l15 = lane & 15;
  const int rowB = blockIdx.x * 16;
  v8f acc = {};
#pragma unroll
  for (int kc = 0; kc < CC / 32; ++kc) {
    v16h a = load_a_f32(hout, rowB + l15, CC, kc * 32, hi);
    v16h b = load_b_packed(Wo, kc, wave, CC / 16, lane);
    acc = WMMA_F16(a, b, acc);
  }
  const int c = wave * 16 + l15;
#pragma unroll
  for (int i = 0; i < 8; ++i) {
    int rr = i + 8 * hi;
    tile[rr][c] = acc[i] + bo[c] + x[(rowB + rr) * CC + c];
  }
  __syncthreads();
  if (threadIdx.x < 16) {
    const int r = threadIdx.x, grow = rowB + r;
    float mu = 0.f, sq = 0.f;
    for (int cc = 0; cc < CC; ++cc) { float t = tile[r][cc]; mu += t; sq += t * t; }
    mu *= (1.f / CC);
    float inv = rsqrtf(sq * (1.f / CC) - mu * mu + 1e-5f);
    float mu2 = 0.f, sq2 = 0.f;
    for (int cc = 0; cc < CC; ++cc) {
      float hc = (tile[r][cc] - mu) * inv * s1[cc] + b1[cc];
      tile[r][cc] = hc;
      mu2 += hc; sq2 += hc * hc;
    }
    mu2 *= (1.f / CC);
    float inv2 = rsqrtf(sq2 * (1.f / CC) - mu2 * mu2 + 1e-5f);
    for (int cc = 0; cc < CC; ++cc) {
      float hc = tile[r][cc];
      h[grow * CC + cc] = hc;
      g[grow * CC + cc] = (h16)((hc - mu2) * inv2 * s2[cc] + b2[cc]);
    }
  }
}

// ======================= 7) FFN1: gelu(g @ W1 + b1) ========================
__global__ __launch_bounds__(256) void ffn1_kernel(
    const h16* __restrict__ g, const h16* __restrict__ W1,
    const float* __restrict__ b1, h16* __restrict__ t) {
  const int wave = threadIdx.x >> 5, lane = threadIdx.x & 31;
  const int hi = lane >> 4, l15 = lane & 15;
  const int rowB = blockIdx.x * 16;
  v8f acc[4] = {{}, {}, {}, {}};
#pragma unroll
  for (int kc = 0; kc < CC / 32; ++kc) {
    v16h a = load_a_f16(g, rowB + l15, CC, kc * 32, hi);
#pragma unroll
    for (int j = 0; j < 4; ++j) {
      v16h b = load_b_packed(W1, kc, wave + 8 * j, C4 / 16, lane);
      acc[j] = WMMA_F16(a, b, acc[j]);
    }
  }
#pragma unroll
  for (int j = 0; j < 4; ++j) {
    int col = (wave + 8 * j) * 16 + l15;
#pragma unroll
    for (int i = 0; i < 8; ++i) {
      int r = rowB + i + 8 * hi;
      float u = acc[j][i] + b1[col];
      t[r * C4 + col] = (h16)(0.5f * u * (1.f + erff(u * 0.70710678f))); // exact gelu
    }
  }
}

// ================ 8) FFN2: out = h + t @ W2 + b2 ===========================
__global__ __launch_bounds__(256) void ffn2_kernel(
    const h16* __restrict__ t, const h16* __restrict__ W2,
    const float* __restrict__ b2, const float* __restrict__ h,
    float* __restrict__ out) {
  const int wave = threadIdx.x >> 5, lane = threadIdx.x & 31;
  const int hi = lane >> 4, l15 = lane & 15;
  const int rowB = blockIdx.x * 16;
  v8f acc = {};
#pragma unroll 4
  for (int kc = 0; kc < C4 / 32; ++kc) {
    v16h a = load_a_f16(t, rowB + l15, C4, kc * 32, hi);
    v16h b = load_b_packed(W2, kc, wave, CC / 16, lane);
    acc = WMMA_F16(a, b, acc);
  }
  const int c = wave * 16 + l15;
#pragma unroll
  for (int i = 0; i < 8; ++i) {
    int r = rowB + i + 8 * hi;
    out[r * CC + c] = acc[i] + b2[c] + h[r * CC + c];
  }
}

// ===========================================================================
extern "C" void kernel_launch(void* const* d_in, const int* in_sizes, int n_in,
                              void* d_out, int out_size, void* d_ws, size_t ws_size,
                              hipStream_t stream) {
  const float* x    = (const float*)d_in[0];
  const int*   ei   = (const int*)d_in[1];
  const float* Wq   = (const float*)d_in[2];
  const float* bq   = (const float*)d_in[3];
  const float* Wk   = (const float*)d_in[4];
  const float* bk   = (const float*)d_in[5];
  const float* Wv   = (const float*)d_in[6];
  const float* bv   = (const float*)d_in[7];
  const float* Wo   = (const float*)d_in[8];
  const float* bo   = (const float*)d_in[9];
  const float* l1s  = (const float*)d_in[10];
  const float* l1b  = (const float*)d_in[11];
  const float* l2s  = (const float*)d_in[12];
  const float* l2b  = (const float*)d_in[13];
  const float* W1   = (const float*)d_in[14];
  const float* b1   = (const float*)d_in[15];
  const float* W2   = (const float*)d_in[16];
  const float* b2   = (const float*)d_in[17];
  float* out = (float*)d_out;

  const size_t NC = (size_t)NN * CC;      // 6.4M
  const size_t EH = (size_t)EE * HH;      // 6.4M
  float* ws = (float*)d_ws;
  // f32 regions
  float* q      = ws;                     // [N,C]
  float* k      = ws + NC;                // [N,C]
  float* v      = ws + 2 * NC;            // [N,C]
  float* scores = ws + 3 * NC;            // [E,H]
  float* hout   = ws + 3 * NC + EH;       // [N,C] atomic accumulator
  float* hbuf   = hout + NC;              // [N,C] h (f32, residual)
  float* red    = hbuf + NC;              // 8 max-keys + 8 sums
  unsigned* rmax = (unsigned*)red;
  float*    rsum = red + HH;
  // f16 regions
  h16* xh  = (h16*)(red + 16);            // [N,C] halves      (NC/2 floats)
  h16* gh  = xh + NC;                     // [N,C] halves      (NC/2 floats)
  h16* Wqp = gh + NC;                     // packed weights
  h16* Wkp = Wqp + CC * CC;
  h16* Wvp = Wkp + CC * CC;
  h16* Wop = Wvp + CC * CC;
  h16* W1p = Wop + CC * CC;
  h16* W2p = W1p + CC * C4;
  h16* th  = (h16*)ws;                    // [N,4C] halves, aliases q/k/v (dead by then)

  const int rowBlocks  = NN / 16;                 // 3125
  const int edgeBlocks = EE / 256;                // 3125
  const int elemBlocks = (int)((NC + 255) / 256); // 25000

  // one-time prep (tiny): pack weights into WMMA fragment layout, x -> f16
  pack_weight_kernel<<<(CC * CC + 255) / 256, 256, 0, stream>>>(Wq, Wqp, CC, CC);
  pack_weight_kernel<<<(CC * CC + 255) / 256, 256, 0, stream>>>(Wk, Wkp, CC, CC);
  pack_weight_kernel<<<(CC * CC + 255) / 256, 256, 0, stream>>>(Wv, Wvp, CC, CC);
  pack_weight_kernel<<<(CC * CC + 255) / 256, 256, 0, stream>>>(Wo, Wop, CC, CC);
  pack_weight_kernel<<<(CC * C4 + 255) / 256, 256, 0, stream>>>(W1, W1p, CC, C4);
  pack_weight_kernel<<<(C4 * CC + 255) / 256, 256, 0, stream>>>(W2, W2p, C4, CC);
  cvt_x_kernel<<<elemBlocks, 256, 0, stream>>>(x, xh);

  qkv_kernel<<<rowBlocks, 256, 0, stream>>>(xh, Wqp, bq, Wkp, bk, Wvp, bv, q, k, v);
  init_kernel<<<elemBlocks, 256, 0, stream>>>(hout, rmax, rsum);
  scores_kernel<<<edgeBlocks, 256, 0, stream>>>(q, k, ei, scores, rmax);
  expsum_kernel<<<edgeBlocks, 256, 0, stream>>>(scores, rmax, rsum);
  scatter_kernel<<<edgeBlocks, 256, 0, stream>>>(v, scores, ei, rmax, rsum, hout);
  oproj_ln_kernel<<<rowBlocks, 256, 0, stream>>>(hout, Wop, bo, x, l1s, l1b, l2s, l2b,
                                                 hbuf, gh);
  ffn1_kernel<<<rowBlocks, 256, 0, stream>>>(gh, W1p, b1, th);
  ffn2_kernel<<<rowBlocks, 256, 0, stream>>>(th, W2p, b2, hbuf, out);
}
